// InterpretableTensorClassifier_82154134438102
// MI455X (gfx1250) — matrix operand order repaired and checked by reading, not compiled, not run on hardware
//
#include <hip/hip_runtime.h>
#include <hip/hip_bf16.h>

#define KD 256
#define BD 64
#define MAT (KD * KD)   // 65536
#define NSWEEP 4

typedef __attribute__((ext_vector_type(16))) unsigned short u16x16;
typedef __attribute__((ext_vector_type(8)))  unsigned short u16x8;
typedef __attribute__((ext_vector_type(16))) __bf16 bf16x16;
typedef __attribute__((ext_vector_type(8)))  float  f32x8;
typedef __attribute__((ext_vector_type(4)))  unsigned int u32x4;
typedef __attribute__((ext_vector_type(8)))  int i32x8;
typedef __attribute__((ext_vector_type(4)))  int i32x4;

// Feature detection for gfx1250 data movers (fallback path always compiles).
#if __has_builtin(__builtin_amdgcn_tensor_load_to_lds) && __has_builtin(__builtin_amdgcn_s_wait_tensorcnt)
#define HAVE_TDM 1
#elif __has_builtin(__builtin_amdgcn_global_load_async_to_lds_b128) && __has_builtin(__builtin_amdgcn_s_wait_asynccnt)
#define HAVE_ASYNC 1
#endif

// ---------------------------------------------------------------- helpers

__device__ __forceinline__ unsigned short f2bf(float f) {
  unsigned int x = __float_as_uint(f);
  x += 0x7FFFu + ((x >> 16) & 1u);   // round-to-nearest-even
  return (unsigned short)(x >> 16);
}

__device__ __forceinline__ f32x8 wmma_bf16(bf16x16 a, bf16x16 b, f32x8 c) {
  return __builtin_amdgcn_wmma_f32_16x16x32_bf16(false, a, false, b,
                                                 (short)0, c, false, false);
}

// A fragment (16x32 bf16, row-major source). Lane L (0-15): row M=L, elems
// 0..7 = K k0..k0+7, elems 8..15 = K k0+16..k0+23 ; lanes 16-31: k0 += 8.
__device__ __forceinline__ bf16x16 fragA_bf(const unsigned short* __restrict__ A,
                                            int tm, int tk, int lrow, int lhi) {
  const unsigned short* p = A + (tm * 16 + lrow) * KD + tk * 32 + (lhi << 3);
  u16x16 u;
#pragma unroll
  for (int i = 0; i < 8; ++i) { u[i] = p[i]; u[8 + i] = p[16 + i]; }
  return __builtin_bit_cast(bf16x16, u);
}

// B fragment (32x16 bf16) from an "effective column-major" buffer in LDS:
// Bcm[n*256 + k] == B[k][n].  Lane L (0-15): col N=L, elems 0..7 = K
// k0..k0+7, elems 8..15 = K k0+16..k0+23 ; lanes 16-31: k0 += 8.
__device__ __forceinline__ bf16x16 fragB_lds(const unsigned short* Bcm,
                                             int tn, int tk, int lcol, int lhi) {
  const unsigned short* p = Bcm + (tn * 16 + lcol) * KD + tk * 32 + (lhi << 3);
  u16x16 u;
#pragma unroll
  for (int i = 0; i < 8; ++i) { u[i] = p[i]; u[8 + i] = p[16 + i]; }
  return __builtin_bit_cast(bf16x16, u);
}

// C tile store: VGPR i -> M = (lane>=16 ? 8 : 0) + i, N = lane&15.
__device__ __forceinline__ void storeC_bf(unsigned short* __restrict__ C,
                                          int tm, int tn, int lane, f32x8 acc) {
  int n = tn * 16 + (lane & 15);
  int m0 = tm * 16 + ((lane >> 4) << 3);
#pragma unroll
  for (int i = 0; i < 8; ++i) C[(m0 + i) * KD + n] = f2bf(acc[i]);
}

__device__ __forceinline__ void storeC_f32(float* __restrict__ C,
                                           int tm, int tn, int lane, f32x8 acc) {
  int n = tn * 16 + (lane & 15);
  int m0 = tm * 16 + ((lane >> 4) << 3);
#pragma unroll
  for (int i = 0; i < 8; ++i) C[(m0 + i) * KD + n] = acc[i];
}

#if defined(HAVE_TDM)
__device__ __forceinline__ unsigned int lds_offset_of(const void* p) {
  return (unsigned int)(size_t)(__attribute__((address_space(3))) const void*)p;
}
#endif

// Stage one 256x256 bf16 matrix (128 KB) global -> LDS for the whole WG.
__device__ __forceinline__ void stage_B(unsigned short* lds,
                                        const unsigned short* gsrc,
                                        int tid, int wave) {
  __threadfence();       // make prior global C-tile stores visible
  __syncthreads();       // all waves done with previous LDS contents
#if defined(HAVE_TDM)
  if (wave == 0) {
    unsigned long long ga = (unsigned long long)(size_t)gsrc;
    u32x4 g0;
    g0[0] = 1u;                                   // count=1, user mode, no gather
    g0[1] = lds_offset_of(lds);                   // lds_addr (bytes)
    g0[2] = (unsigned int)(ga & 0xFFFFFFFFu);     // global_addr[31:0]
    g0[3] = (unsigned int)((ga >> 32) & 0x01FFFFFFu) | 0x80000000u; // addr[56:32] | type=2
    i32x8 g1;
    g1[0] = 0x00010000;   // workgroup_mask=0, data_size=1 (2 bytes)
    g1[1] = 0x01000000;   // tensor_dim0 = 256 (bits 63:48)
    g1[2] = 0x01000000;   // tensor_dim1 = 256 (bits 95:80)
    g1[3] = 0x01000000;   // tile_dim0   = 256 (bits 127:112)
    g1[4] = 0x00000100;   // tile_dim1   = 256 (bits 143:128)
    g1[5] = 256;          // tensor_dim0_stride = 256 elements
    g1[6] = 0;
    g1[7] = 0;
    i32x4 gz = {0, 0, 0, 0};
#if __clang_major__ >= 23
    i32x8 gz8 = {0, 0, 0, 0, 0, 0, 0, 0};
    __builtin_amdgcn_tensor_load_to_lds(g0, g1, gz, gz, gz8, 0);
#else
    __builtin_amdgcn_tensor_load_to_lds(g0, g1, gz, gz, 0);
#endif
    __builtin_amdgcn_s_wait_tensorcnt(0);
  }
  __syncthreads();
#elif defined(HAVE_ASYNC)
  for (int i = tid; i < 8192; i += 256) {
    __builtin_amdgcn_global_load_async_to_lds_b128(
        (const __attribute__((address_space(1))) void*)(gsrc + i * 8),
        (__attribute__((address_space(3))) void*)(lds + i * 8), 0, 0);
  }
  __builtin_amdgcn_s_wait_asynccnt(0);
  __syncthreads();
#else
  for (int i = tid; i < 8192; i += 256) {
    *(u16x8*)(lds + i * 8) = *(const u16x8*)(gsrc + i * 8);
  }
  __syncthreads();
#endif
}

// ---------------------------------------------------------------- conversion

__global__ __launch_bounds__(256) void cvt_zr_cm_kernel(
    const float* __restrict__ tin, unsigned short* __restrict__ dst) {
  int idx = blockIdx.x * 256 + threadIdx.x;   // 2*64*65536 threads
  int m = idx >> 16;                          // chain*64 + b
  int p = idx & 65535;
  int chain = m >> 6, b = m & 63;
  int n = p >> 8, k = p & 255;
  dst[idx] = f2bf(tin[((size_t)b * 6 + 4 + chain) * MAT + k * KD + n]);
}

// Row-major bf16 copies: a_rm = ic1/sir1 (chain A0), b3_rm = sir2/ic2 (last B,
// used transposed, so its row-major layout IS the effective column-major).
__global__ __launch_bounds__(256) void cvt_rm_kernel(
    const float* __restrict__ tin,
    unsigned short* __restrict__ a_rm, unsigned short* __restrict__ b3_rm) {
  int idx = blockIdx.x * 256 + threadIdx.x;   // 2*64*65536 threads
  int m = idx >> 16;
  int p = idx & 65535;
  int chain = m >> 6, b = m & 63;
  a_rm[idx]  = f2bf(tin[((size_t)b * 6 + (chain ? 2 : 0)) * MAT + p]);
  b3_rm[idx] = f2bf(tin[((size_t)b * 6 + (chain ? 1 : 3)) * MAT + p]);
}

__global__ __launch_bounds__(256) void cvt_s_cm_kernel(
    const float* __restrict__ S, unsigned short* __restrict__ dst) {
  int idx = blockIdx.x * 256 + threadIdx.x;   // 65536 threads
  int n = idx >> 8, k = idx & 255;
  dst[idx] = f2bf(S[k * KD + n]);
}

// ---------------------------------------------------------------- GEMM chain
// One workgroup (8 wave32) per (chain,b):
//   chain0: Z1 = (((ic1*S1)*Z1r)*S2)*sir2^T
//   chain1: Z2 = (((sir1*S3)*Z2r)*S4)*ic2^T
// B operand of each step staged into LDS (TDM/async/copy), 16x reuse per wave.
__global__ __launch_bounds__(256) void chain_gemm_kernel(
    const unsigned short* __restrict__ a_rm,
    const unsigned short* __restrict__ b3_rm,
    const unsigned short* __restrict__ zr_cm,
    const unsigned short* __restrict__ s_cm,
    unsigned short* __restrict__ T0,
    unsigned short* __restrict__ T1,
    float* __restrict__ Z) {
  extern __shared__ unsigned short ldsB[];   // 65536 u16 = 128 KB (dynamic)
  const int m = blockIdx.x;        // 0..127 = chain*64 + b
  const int chain = m >> 6;
  const int tid = threadIdx.x, wave = tid >> 5, lane = tid & 31;
  const int lrow = lane & 15, lhi = lane >> 4;

  const unsigned short* A0  = a_rm  + (size_t)m * MAT;
  const unsigned short* B3  = b3_rm + (size_t)m * MAT;
  const unsigned short* Bs1 = s_cm + (size_t)(chain ? 2 : 0) * MAT;  // S1 / S3
  const unsigned short* Bs2 = s_cm + (size_t)(chain ? 3 : 1) * MAT;  // S2 / S4
  const unsigned short* Bzr = zr_cm + (size_t)m * MAT;               // Z1r / Z2r
  unsigned short* t0 = T0 + (size_t)m * MAT;
  unsigned short* t1 = T1 + (size_t)m * MAT;
  float* zout = Z + (size_t)m * MAT;

  // ---- step 0: t0 = A0 @ S
  stage_B(ldsB, Bs1, tid, wave);
  for (int tile = wave; tile < 256; tile += 8) {
    int tm = tile >> 4, tn = tile & 15;
    f32x8 acc = {};
#pragma unroll
    for (int tk = 0; tk < 8; ++tk) {
      bf16x16 a = fragA_bf(A0, tm, tk, lrow, lhi);
      bf16x16 bb = fragB_lds(ldsB, tn, tk, lrow, lhi);
      acc = wmma_bf16(a, bb, acc);
    }
    storeC_bf(t0, tm, tn, lane, acc);
  }

  // ---- step 1: t1 = t0 @ Zr
  stage_B(ldsB, Bzr, tid, wave);
  for (int tile = wave; tile < 256; tile += 8) {
    int tm = tile >> 4, tn = tile & 15;
    __builtin_prefetch(t0 + (tm * 16 + lrow) * KD, 0, 1);
    f32x8 acc = {};
#pragma unroll
    for (int tk = 0; tk < 8; ++tk) {
      bf16x16 a = fragA_bf(t0, tm, tk, lrow, lhi);
      bf16x16 bb = fragB_lds(ldsB, tn, tk, lrow, lhi);
      acc = wmma_bf16(a, bb, acc);
    }
    storeC_bf(t1, tm, tn, lane, acc);
  }

  // ---- step 2: t0 = t1 @ S'
  stage_B(ldsB, Bs2, tid, wave);
  for (int tile = wave; tile < 256; tile += 8) {
    int tm = tile >> 4, tn = tile & 15;
    __builtin_prefetch(t1 + (tm * 16 + lrow) * KD, 0, 1);
    f32x8 acc = {};
#pragma unroll
    for (int tk = 0; tk < 8; ++tk) {
      bf16x16 a = fragA_bf(t1, tm, tk, lrow, lhi);
      bf16x16 bb = fragB_lds(ldsB, tn, tk, lrow, lhi);
      acc = wmma_bf16(a, bb, acc);
    }
    storeC_bf(t0, tm, tn, lane, acc);
  }

  // ---- step 3: Z = t0 @ Mt^T
  stage_B(ldsB, B3, tid, wave);
  for (int tile = wave; tile < 256; tile += 8) {
    int tm = tile >> 4, tn = tile & 15;
    f32x8 acc = {};
#pragma unroll
    for (int tk = 0; tk < 8; ++tk) {
      bf16x16 a = fragA_bf(t0, tm, tk, lrow, lhi);
      bf16x16 bb = fragB_lds(ldsB, tn, tk, lrow, lhi);
      acc = wmma_bf16(a, bb, acc);
    }
    storeC_f32(zout, tm, tn, lane, acc);
  }
}

// ---------------------------------------------------------------- reductions

__device__ __forceinline__ float block_reduce(float v, float* red, int tid) {
  red[tid] = v; __syncthreads();
  for (int s = 128; s > 0; s >>= 1) { if (tid < s) red[tid] += red[tid + s]; __syncthreads(); }
  float r = red[0]; __syncthreads();
  return r;
}

__global__ __launch_bounds__(256) void stats_z_kernel(
    const float* __restrict__ Z, const float* __restrict__ tin,
    float* __restrict__ diffm, float* __restrict__ offm, float* __restrict__ trm) {
  __shared__ float red[256];
  int m = blockIdx.x, chain = m >> 6, b = m & 63, tid = threadIdx.x;
  const float* G = Z + (size_t)m * MAT;
  const float* Zr = tin + ((size_t)b * 6 + 4 + chain) * MAT;
  float sd = 0.f, so = 0.f, st = 0.f;
  for (int idx = tid; idx < MAT; idx += 256) {
    int i = idx >> 8, j = idx & 255;
    float z = G[idx];
    sd += fabsf(z - (i == j ? 1.0f : 0.0f));
    if (i != j) so += fabsf(z);
    st += fabsf(z - Zr[idx]);
  }
  float td = block_reduce(sd, red, tid);
  float to = block_reduce(so, red, tid);
  float tt = block_reduce(st, red, tid);
  if (tid == 0) {
    diffm[m] = td / 65536.0f;
    offm[m]  = to / 65536.0f;
    trm[m]   = tt / 65536.0f;
  }
}

__global__ __launch_bounds__(256) void sreg_kernel(
    const float* __restrict__ S0, const float* __restrict__ S1,
    const float* __restrict__ S2, const float* __restrict__ S3,
    float* __restrict__ sregA, float* __restrict__ sregB) {
  __shared__ float red[256];
  int tid = threadIdx.x;
  const float* S = (blockIdx.x == 0) ? S0 : (blockIdx.x == 1) ? S1
                  : (blockIdx.x == 2) ? S2 : S3;
  float sa = 0.f, sb = 0.f;
  for (int idx = tid; idx < MAT; idx += 256) {
    int i = idx >> 8, j = idx & 255;
    float v = S[idx];
    sa += fabsf(v - (i == j ? 1.0f : 0.0f));
    if (i != j) sb += fabsf(v);
  }
  float ta = block_reduce(sa, red, tid);
  float tb = block_reduce(sb, red, tid);
  if (tid == 0) { sregA[blockIdx.x] = ta / 65536.0f; sregB[blockIdx.x] = tb / 65536.0f; }
}

// CP reconstruction MSE partials: one block per (b,i,j), thread = k.
__global__ __launch_bounds__(256) void cp_kernel(
    const float* __restrict__ tin, const float* __restrict__ Z,
    const float* __restrict__ cpA, const float* __restrict__ cpB,
    const float* __restrict__ cpC, float* __restrict__ partial) {
  __shared__ float red[256];
  int blk = blockIdx.x;                  // 64*6*256 blocks
  int b = blk / (6 * 256);
  int rem = blk % (6 * 256);
  int i = rem >> 8, j = rem & 255;
  int k = threadIdx.x;
  float w0 = cpA[(b * 6 + i) * 4 + 0] * cpB[(b * 256 + j) * 4 + 0];
  float w1 = cpA[(b * 6 + i) * 4 + 1] * cpB[(b * 256 + j) * 4 + 1];
  float w2 = cpA[(b * 6 + i) * 4 + 2] * cpB[(b * 256 + j) * 4 + 2];
  float w3 = cpA[(b * 6 + i) * 4 + 3] * cpB[(b * 256 + j) * 4 + 3];
  float recon = w0 * cpC[(b * 256 + k) * 4 + 0] + w1 * cpC[(b * 256 + k) * 4 + 1]
              + w2 * cpC[(b * 256 + k) * 4 + 2] + w3 * cpC[(b * 256 + k) * 4 + 3];
  float t;
  if (i < 4) t = tin[(((size_t)b * 6 + i) * KD + j) * KD + k];
  else       t = Z[(((size_t)(i - 4) * 64 + b) * KD + j) * KD + k];
  float d = recon - t;
  float s = block_reduce(d * d, red, threadIdx.x);
  if (threadIdx.x == 0) partial[blk] = s;
}

// ---------------------------------------------------------------- eigvalsh
// Parallel cyclic Jacobi, in-place on Z (run after all raw-Z consumers).
__global__ __launch_bounds__(256) void jacobi_kernel(
    float* __restrict__ Z,
    const float* __restrict__ val_out, const float* __restrict__ val_in,
    const float* __restrict__ diffm, const float* __restrict__ offm,
    float* __restrict__ lossm) {
  int m = blockIdx.x, chain = m >> 6, b = m & 63, tid = threadIdx.x;
  float* G = Z + (size_t)m * MAT;

  // symmetrize using lower triangle (eigvalsh UPLO='L')
  for (int idx = tid; idx < MAT; idx += 256) {
    int i = idx >> 8, j = idx & 255;
    if (i < j) G[idx] = G[j * KD + i];
  }
  __threadfence(); __syncthreads();

  __shared__ float cs[128], sn[128];
  __shared__ int pp[128], qq[128];

  for (int sweep = 0; sweep < NSWEEP; ++sweep) {
    for (int r = 0; r < 255; ++r) {
      if (tid < 128) {
        int p, q;
        if (tid == 0) { p = 255; q = r; }
        else { p = (r + tid) % 255; q = (r + 255 - tid) % 255; }
        float app = G[p * KD + p], aqq = G[q * KD + q], apq = G[p * KD + q];
        float c = 1.0f, s = 0.0f;
        if (fabsf(apq) > 1e-20f) {
          float theta = (aqq - app) / (2.0f * apq);
          float t = copysignf(1.0f, theta) / (fabsf(theta) + sqrtf(1.0f + theta * theta));
          c = rsqrtf(1.0f + t * t);
          s = t * c;
        }
        cs[tid] = c; sn[tid] = s; pp[tid] = p; qq[tid] = q;
      }
      __syncthreads();
      {  // row phase: rows p,q <- J^T A
        int pr = tid >> 1, j0 = (tid & 1) * 128;
        int p = pp[pr], q = qq[pr];
        float c = cs[pr], s = sn[pr];
        for (int j = j0; j < j0 + 128; ++j) {
          float gp = G[p * KD + j], gq = G[q * KD + j];
          G[p * KD + j] = c * gp - s * gq;
          G[q * KD + j] = s * gp + c * gq;
        }
      }
      __threadfence(); __syncthreads();
      {  // col phase: cols p,q <- B J
        int pr = tid >> 1, i0 = (tid & 1) * 128;
        int p = pp[pr], q = qq[pr];
        float c = cs[pr], s = sn[pr];
        for (int i = i0; i < i0 + 128; ++i) {
          float gp = G[i * KD + p], gq = G[i * KD + q];
          G[i * KD + p] = c * gp - s * gq;
          G[i * KD + q] = s * gp + c * gq;
        }
      }
      __threadfence(); __syncthreads();
    }
  }

  __shared__ float dv[256], ev[256];
  dv[tid] = G[tid * KD + tid];
  __syncthreads();
  float v = dv[tid];
  int rank = 0;
  for (int j = 0; j < 256; ++j) {
    float o = dv[j];
    rank += (o < v) || (o == v && j < tid);
  }
  ev[rank] = v;
  __syncthreads();
  if (tid == 0) {
    float val = (chain == 0 ? val_out : val_in)[b];
    float el = 0.25f * (fabsf(val - ev[1]) + fabsf(val - ev[2]) +
                        fabsf(val - ev[3]) + fabsf(val - ev[4]));
    float loss = el + 0.2f * diffm[m] + 0.1f * (-offm[m]);
    lossm[m] = (val != 0.0f) ? loss : 0.0f;
  }
}

// ---------------------------------------------------------------- MLP heads

__global__ __launch_bounds__(64) void heads_kernel(
    const float* __restrict__ cpA, const float* __restrict__ cpB,
    const float* __restrict__ cpC, const float* __restrict__ cpW1,
    const float* __restrict__ cpb1, const float* __restrict__ cpw2,
    const float* __restrict__ cpb2,
    const float* __restrict__ seed_ic, const float* __restrict__ seed_sir,
    const float* __restrict__ seed_icc, const float* __restrict__ seed_sirc,
    const float* __restrict__ seedW1, const float* __restrict__ seedb1,
    const float* __restrict__ seedw2, const float* __restrict__ seedb2,
    const float* __restrict__ scalars,
    const float* __restrict__ finW1, const float* __restrict__ finb1,
    const float* __restrict__ finw2, const float* __restrict__ finb2,
    float* __restrict__ out) {
  int b = threadIdx.x;
  if (b >= 64) return;

  float cps[4];
  for (int r = 0; r < 4; ++r) {
    float d0 = 0.f, d1 = 0.f, d2 = 0.f;
    for (int i = 0; i < 6; ++i) d0 += cpA[(b * 6 + i) * 4 + r];
    d0 *= (1.0f / 6.0f);
    for (int j = 0; j < 256; ++j) d1 += cpB[(b * 256 + j) * 4 + r];
    d1 *= (1.0f / 256.0f);
    for (int k = 0; k < 256; ++k) d2 += cpC[(b * 256 + k) * 4 + r];
    d2 *= (1.0f / 256.0f);
    float s = 0.f;
    for (int h = 0; h < 32; ++h) {
      float z = cpb1[r * 32 + h]
              + d0 * cpW1[(r * 32 + h) * 3 + 0]
              + d1 * cpW1[(r * 32 + h) * 3 + 1]
              + d2 * cpW1[(r * 32 + h) * 3 + 2];
      s += fmaxf(z, 0.0f) * cpw2[r * 32 + h];
    }
    cps[r] = s + cpb2[r];
  }

  const float* seeds[4] = {seed_ic, seed_sir, seed_icc, seed_sirc};
  float sds[4];
  for (int j = 0; j < 4; ++j) {
    float s = 0.f;
    for (int h = 0; h < 32; ++h) {
      float z = seedb1[j * 32 + h];
      for (int k = 0; k < 256; ++k)
        z += seeds[j][b * 256 + k] * seedW1[(j * 32 + h) * 256 + k];
      s += fmaxf(z, 0.0f) * seedw2[j * 32 + h];
    }
    sds[j] = s + seedb2[j];
  }

  float comb[10] = {cps[0], cps[1], cps[2], cps[3],
                    sds[0], sds[1], sds[2], sds[3],
                    scalars[b * 2 + 0], scalars[b * 2 + 1]};
  float logit = finb2[0];
  for (int h = 0; h < 32; ++h) {
    float z = finb1[h];
    for (int d = 0; d < 10; ++d) z += comb[d] * finW1[h * 10 + d];
    logit += fmaxf(z, 0.0f) * finw2[h];
  }
  out[b] = logit;
}

__global__ __launch_bounds__(256) void finalize_kernel(
    const float* __restrict__ lossm, const float* __restrict__ sregA,
    const float* __restrict__ sregB, const float* __restrict__ trm,
    const float* __restrict__ cp_partial, int ncp, float* __restrict__ out) {
  __shared__ float red[256];
  int tid = threadIdx.x;
  float s = 0.f;
  for (int i = tid; i < ncp; i += 256) s += cp_partial[i];
  float tot = block_reduce(s, red, tid);
  if (tid == 0) {
    float cp_loss = tot / (64.0f * 6.0f * 65536.0f);
    float so = 0.f, si = 0.f, tr = 0.f;
    for (int b = 0; b < 64; ++b) {
      so += lossm[b];
      si += lossm[64 + b];
      tr += trm[b] + trm[64 + b];
    }
    so *= (1.0f / 64.0f); si *= (1.0f / 64.0f); tr *= (1.0f / 64.0f);
    float st = 0.f;
    for (int i = 0; i < 4; ++i) st += -sregA[i] - sregB[i];
    out[64] = so;
    out[65] = si;
    out[66] = cp_loss;
    out[67] = 0.5f * st + 0.5f * tr;
  }
}

// ---------------------------------------------------------------- launch

extern "C" void kernel_launch(void* const* d_in, const int* in_sizes, int n_in,
                              void* d_out, int out_size, void* d_ws, size_t ws_size,
                              hipStream_t stream) {
  const float* tin       = (const float*)d_in[0];
  const float* seed_ic   = (const float*)d_in[1];
  const float* seed_sir  = (const float*)d_in[2];
  const float* seed_icc  = (const float*)d_in[3];
  const float* seed_sirc = (const float*)d_in[4];
  const float* scalars   = (const float*)d_in[5];
  const float* val_out   = (const float*)d_in[6];
  const float* val_in    = (const float*)d_in[7];
  const float* S1 = (const float*)d_in[8];
  const float* S2 = (const float*)d_in[9];
  const float* S3 = (const float*)d_in[10];
  const float* S4 = (const float*)d_in[11];
  const float* cpA = (const float*)d_in[12];
  const float* cpB = (const float*)d_in[13];
  const float* cpC = (const float*)d_in[14];
  const float* cpW1 = (const float*)d_in[15];
  const float* cpb1 = (const float*)d_in[16];
  const float* cpw2 = (const float*)d_in[17];
  const float* cpb2 = (const float*)d_in[18];
  const float* seedW1 = (const float*)d_in[19];
  const float* seedb1 = (const float*)d_in[20];
  const float* seedw2 = (const float*)d_in[21];
  const float* seedb2 = (const float*)d_in[22];
  const float* finW1 = (const float*)d_in[23];
  const float* finb1 = (const float*)d_in[24];
  const float* finw2 = (const float*)d_in[25];
  const float* finb2 = (const float*)d_in[26];
  float* out = (float*)d_out;

  char* ws = (char*)d_ws;
  size_t off = 0;
  auto carve = [&](size_t bytes) -> char* {
    char* p = ws + off;
    off = (off + bytes + 255) & ~(size_t)255;
    return p;
  };
  unsigned short* zr_cm = (unsigned short*)carve(2ull * BD * MAT * 2);  // 16 MB
  unsigned short* a_rm  = (unsigned short*)carve(2ull * BD * MAT * 2);  // 16 MB
  unsigned short* b3_rm = (unsigned short*)carve(2ull * BD * MAT * 2);  // 16 MB
  unsigned short* s_cm  = (unsigned short*)carve(4ull * MAT * 2);       // 0.5 MB
  unsigned short* T0    = (unsigned short*)carve(2ull * BD * MAT * 2);  // 16 MB
  unsigned short* T1    = (unsigned short*)carve(2ull * BD * MAT * 2);  // 16 MB
  float* Z     = (float*)carve(2ull * BD * MAT * 4);                    // 32 MB
  float* diffm = (float*)carve(128 * 4);
  float* offm  = (float*)carve(128 * 4);
  float* trm   = (float*)carve(128 * 4);
  float* lossm = (float*)carve(128 * 4);
  float* sregA = (float*)carve(4 * 4);
  float* sregB = (float*)carve(4 * 4);
  const int NCP = 64 * 6 * 256;
  float* cp_partial = (float*)carve((size_t)NCP * 4);

  cvt_zr_cm_kernel<<<(2 * BD * MAT) / 256, 256, 0, stream>>>(tin, zr_cm);
  cvt_rm_kernel<<<(2 * BD * MAT) / 256, 256, 0, stream>>>(tin, a_rm, b3_rm);
  cvt_s_cm_kernel<<<MAT / 256, 256, 0, stream>>>(S1, s_cm + 0 * (size_t)MAT);
  cvt_s_cm_kernel<<<MAT / 256, 256, 0, stream>>>(S2, s_cm + 1 * (size_t)MAT);
  cvt_s_cm_kernel<<<MAT / 256, 256, 0, stream>>>(S3, s_cm + 2 * (size_t)MAT);
  cvt_s_cm_kernel<<<MAT / 256, 256, 0, stream>>>(S4, s_cm + 3 * (size_t)MAT);

  chain_gemm_kernel<<<128, 256, 131072, stream>>>(a_rm, b3_rm, zr_cm, s_cm,
                                                  T0, T1, Z);

  stats_z_kernel<<<128, 256, 0, stream>>>(Z, tin, diffm, offm, trm);
  sreg_kernel<<<4, 256, 0, stream>>>(S1, S2, S3, S4, sregA, sregB);
  cp_kernel<<<NCP, 256, 0, stream>>>(tin, Z, cpA, cpB, cpC, cp_partial);

  jacobi_kernel<<<128, 256, 0, stream>>>(Z, val_out, val_in, diffm, offm, lossm);

  heads_kernel<<<1, 64, 0, stream>>>(cpA, cpB, cpC, cpW1, cpb1, cpw2, cpb2,
                                     seed_ic, seed_sir, seed_icc, seed_sirc,
                                     seedW1, seedb1, seedw2, seedb2,
                                     scalars, finW1, finb1, finw2, finb2, out);
  finalize_kernel<<<1, 256, 0, stream>>>(lossm, sregA, sregB, trm,
                                         cp_partial, NCP, out);
}
